// GraphTransformer_19713899889539
// MI455X (gfx1250) — matrix operand order repaired and checked by reading, not compiled, not run on hardware
//
#include <hip/hip_runtime.h>
#include <hip/hip_bf16.h>
#include <float.h>

// ---------------------------------------------------------------------------
// GAT (GraphTransformer) for MI455X / gfx1250.
//  - GEMM: pre-packed bf16 WMMA fragments -> v_wmma_f32_16x16x32_bf16,
//    2x global_load_b128 per operand per k-step, zero guards in hot loop.
//  - segment softmax via ordered-uint atomicMax + f32 atomicAdd scatter.
// ---------------------------------------------------------------------------

#define N_NODES 50000
#define N_EDGES 600000
#define BN_EPS  1e-5f
#define NEG_SLOPE 0.2f

typedef __attribute__((ext_vector_type(16))) __bf16 v16bf;
typedef __attribute__((ext_vector_type(8)))  float  v8f;

// ---- helpers --------------------------------------------------------------

__device__ inline __bf16 to_bf16(float f) {
  unsigned b = __float_as_uint(f);
  unsigned r = (b + 0x7FFFu + ((b >> 16) & 1u)) >> 16;   // RNE
  unsigned short s = (unsigned short)r;
  return __builtin_bit_cast(__bf16, s);
}

// Order-preserving float <-> uint map (atomicMax with negative floats).
__device__ inline unsigned fenc(float x) {
  unsigned b = __float_as_uint(x);
  return b ^ ((b >> 31) ? 0xFFFFFFFFu : 0x80000000u);
}
__device__ inline float fdec(unsigned u) {
  return __uint_as_float((u >> 31) ? (u ^ 0x80000000u) : ~u);
}

__device__ inline float leaky(float v) { return v > 0.0f ? v : v * NEG_SLOPE; }

// ---- kernels --------------------------------------------------------------

__global__ void fill_u32(unsigned* __restrict__ p, unsigned v, int n) {
  int i = blockIdx.x * blockDim.x + threadIdx.x;
  if (i < n) p[i] = v;
}

// Pack x[M,Ksrc] (f32) into A fragments (bf16), zero-padding K to nkb*32.
// Fragment layout: [(rb*nkb + kb)*32 + lane]*16 + e
//   row = rb*16 + (lane&15); k = kb*32 + (e&7) + ((e>>3)<<4) + ((lane>>4)<<3)
__global__ void pack_a(const float* __restrict__ x, __bf16* __restrict__ Ap,
                       int Ksrc, int nkb, int total) {
  int i = blockIdx.x * blockDim.x + threadIdx.x;
  if (i >= total) return;
  int e  = i & 15;
  int l  = (i >> 4) & 31;
  int t  = i >> 9;            // rb*nkb + kb
  int kb = t % nkb;
  int rb = t / nkb;
  int row = rb * 16 + (l & 15);
  int k = kb * 32 + (e & 7) + ((e >> 3) << 4) + ((l >> 4) << 3);
  float v = (k < Ksrc) ? x[(size_t)row * Ksrc + k] : 0.0f;
  Ap[i] = to_bf16(v);
}

// Pack W[K,Nc] (f32) into B fragments (bf16), zero-padding K to nkb*32.
// Fragment layout: [(tile*nkb + kb)*32 + lane]*16 + e
//   col = tile*16 + (lane&15); k = kb*32 + (e&7) + ((e>>3)<<4) + ((lane>>4)<<3)
__global__ void pack_b(const float* __restrict__ W, __bf16* __restrict__ Bp,
                       int Ksrc, int Nc, int nkb, int total) {
  int i = blockIdx.x * blockDim.x + threadIdx.x;
  if (i >= total) return;
  int e  = i & 15;
  int l  = (i >> 4) & 31;
  int t  = i >> 9;            // tile*nkb + kb
  int kb = t % nkb;
  int tile = t / nkb;
  int col = tile * 16 + (l & 15);
  int k = kb * 32 + (e & 7) + ((e >> 3) << 4) + ((l >> 4) << 3);
  float v = (k < Ksrc) ? W[(size_t)k * Nc + col] : 0.0f;
  Bp[i] = to_bf16(v);
}

// C[M,Nc] = A*B from packed bf16 fragments. One wave per 16x16 C tile.
// grid.x = M/16 row-blocks; blockDim.x = (Nc/16)*32 (wave w -> column tile w).
template <int NKB>
__global__ void gemm_packed(const __bf16* __restrict__ Ap,
                            const __bf16* __restrict__ Bp,
                            float* __restrict__ C, int Nc) {
  const int rb   = blockIdx.x;
  const int wave = threadIdx.x >> 5;
  const int lane = threadIdx.x & 31;

  const v16bf* ap = (const v16bf*)(Ap + (((size_t)rb * NKB) * 32 + lane) * 16);
  const v16bf* bp = (const v16bf*)(Bp + (((size_t)wave * NKB) * 32 + lane) * 16);

  v8f acc = {};
#pragma unroll
  for (int kb = 0; kb < NKB; ++kb) {
    v16bf a = ap[kb * 32];     // 32 bytes -> 2x global_load_b128
    v16bf b = bp[kb * 32];
    acc = __builtin_amdgcn_wmma_f32_16x16x32_bf16(
        false, a, false, b, (short)0, acc, false, false);
  }

  const int half = lane >> 4;
  const int col  = wave * 16 + (lane & 15);
#pragma unroll
  for (int v = 0; v < 8; ++v) {
    int row = rb * 16 + v + (half << 3);
    C[(size_t)row * Nc + col] = acc[v];
  }
}

// Per-node attention logits: as[n,h] = <hx[n,h,:], a_src[h,:]>, ad likewise.
__global__ void attn_scores(const float* __restrict__ hx,
                            const float* __restrict__ a_src,
                            const float* __restrict__ a_dst,
                            float* __restrict__ as_, float* __restrict__ ad_,
                            int N, int hShift) {
  int i = blockIdx.x * blockDim.x + threadIdx.x;
  int total = N << hShift;
  if (i >= total) return;
  int h = i & ((1 << hShift) - 1);
  int n = i >> hShift;
  const float* row = hx + ((size_t)n << (hShift + 5)) + (h << 5);
  const float* ws = a_src + (h << 5);
  const float* wd = a_dst + (h << 5);
  float s = 0.0f, d = 0.0f;
#pragma unroll
  for (int c = 0; c < 32; ++c) {
    float v = row[c];
    s = fmaf(v, ws[c], s);
    d = fmaf(v, wd[c], d);
  }
  as_[i] = s;
  ad_[i] = d;
}

// Pass 1: segment max of leaky_relu(as[src]+ad[dst]) into menc[dst,h].
__global__ void edge_max(const int* __restrict__ src, const int* __restrict__ dst,
                         const float* __restrict__ as_, const float* __restrict__ ad_,
                         unsigned* __restrict__ menc,
                         int E, int N, int hShift) {
  int i = blockIdx.x * blockDim.x + threadIdx.x;
  int total = (E + N) << hShift;
  if (i >= total) return;
  int e = i >> hShift;
  int h = i & ((1 << hShift) - 1);
  int s = (e < E) ? src[e] : (e - E);
  int d = (e < E) ? dst[e] : (e - E);
  float v = leaky(as_[(s << hShift) + h] + ad_[(d << hShift) + h]);
  atomicMax(&menc[(d << hShift) + h], fenc(v));
}

// Pass 2: ex = exp(v - max[dst]); stash per-edge, accumulate denom[dst,h].
__global__ void edge_exp(const int* __restrict__ src, const int* __restrict__ dst,
                         const float* __restrict__ as_, const float* __restrict__ ad_,
                         const unsigned* __restrict__ menc,
                         float* __restrict__ exbuf, float* __restrict__ denom,
                         int E, int N, int hShift) {
  int i = blockIdx.x * blockDim.x + threadIdx.x;
  int total = (E + N) << hShift;
  if (i >= total) return;
  int e = i >> hShift;
  int h = i & ((1 << hShift) - 1);
  int s = (e < E) ? src[e] : (e - E);
  int d = (e < E) ? dst[e] : (e - E);
  float v = leaky(as_[(s << hShift) + h] + ad_[(d << hShift) + h]);
  float m = fdec(menc[(d << hShift) + h]);
  float ex = __expf(v - m);
  exbuf[i] = ex;
  atomicAdd(&denom[(d << hShift) + h], ex);
}

// Pass 3: normalize in place: alpha = ex / denom[dst,h].
__global__ void edge_alpha(const int* __restrict__ dst,
                           float* __restrict__ exbuf,
                           const float* __restrict__ denom,
                           int E, int N, int hShift) {
  int i = blockIdx.x * blockDim.x + threadIdx.x;
  int total = (E + N) << hShift;
  if (i >= total) return;
  int e = i >> hShift;
  int h = i & ((1 << hShift) - 1);
  int d = (e < E) ? dst[e] : (e - E);
  exbuf[i] = exbuf[i] / denom[(d << hShift) + h];
}

// Pass 4: out[dst, r..r+3] += alpha * hx[src, r..r+3]; 4 channels per thread.
__global__ void edge_agg(const int* __restrict__ src, const int* __restrict__ dst,
                         const float* __restrict__ alpha,
                         const float* __restrict__ hx,
                         float* __restrict__ out,
                         int E, int N, int fShift) {
  int i = blockIdx.x * blockDim.x + threadIdx.x;
  int qShift = fShift - 2;                 // quads per row
  int total = (E + N) << qShift;
  if (i >= total) return;
  int e = i >> qShift;
  int r = (i & ((1 << qShift) - 1)) << 2;  // channel quad start
  int hShift = fShift - 5;
  int h = r >> 5;
  int s = (e < E) ? src[e] : (e - E);
  int d = (e < E) ? dst[e] : (e - E);
  float a = alpha[(e << hShift) + h];
  const float4 v = *(const float4*)(hx + ((size_t)s << fShift) + r);
  float* o = out + ((size_t)d << fShift) + r;
  atomicAdd(o + 0, a * v.x);
  atomicAdd(o + 1, a * v.y);
  atomicAdd(o + 2, a * v.z);
  atomicAdd(o + 3, a * v.w);
}

// Pass 5: x_next = BN_eval(out + bias) [, relu].
__global__ void finalize(const float* __restrict__ out,
                         const float* __restrict__ bias,
                         const float* __restrict__ gamma,
                         const float* __restrict__ beta,
                         const float* __restrict__ rm,
                         const float* __restrict__ rv,
                         float* __restrict__ xnext,
                         int N, int fShift, int do_relu) {
  int i = blockIdx.x * blockDim.x + threadIdx.x;
  int total = N << fShift;
  if (i >= total) return;
  int f = i & ((1 << fShift) - 1);
  float v = out[i] + bias[f];
  v = gamma[f] * (v - rm[f]) * rsqrtf(rv[f] + BN_EPS) + beta[f];
  if (do_relu) v = fmaxf(v, 0.0f);
  xnext[i] = v;
}

// ---- host orchestration ---------------------------------------------------

extern "C" void kernel_launch(void* const* d_in, const int* in_sizes, int n_in,
                              void* d_out, int out_size, void* d_ws, size_t ws_size,
                              hipStream_t stream) {
  (void)in_sizes; (void)n_in; (void)out_size; (void)ws_size;
  const int N = N_NODES, E = N_EDGES;

  const float* x0  = (const float*)d_in[0];
  const int*   src = (const int*)d_in[1];
  const int*   dst = src + E;

  // Workspace carve-out (256B aligned slices).
  char* base = (char*)d_ws;
  auto carve = [&](size_t bytes) {
    char* p = base;
    base += (bytes + 255) & ~(size_t)255;
    return p;
  };
  float*    xbuf  = (float*)carve((size_t)N * 128 * 4);       // layer features
  float*    hx    = (float*)carve((size_t)N * 128 * 4);       // GEMM output
  float*    outb  = (float*)carve((size_t)N * 128 * 4);       // aggregation acc
  __bf16*   Ap    = (__bf16*)carve((size_t)N * 128 * 2);      // packed A frags
  __bf16*   Bp    = (__bf16*)carve((size_t)128 * 128 * 2);    // packed B frags
  float*    exbuf = (float*)carve((size_t)(E + N) * 4 * 4);   // per-edge exp/alpha
  float*    as_   = (float*)carve((size_t)N * 4 * 4);
  float*    ad_   = (float*)carve((size_t)N * 4 * 4);
  unsigned* menc  = (unsigned*)carve((size_t)N * 4 * 4);
  float*    denom = (float*)carve((size_t)N * 4 * 4);

  const int fis[4] = {74, 128, 128, 128};
  const int Hs[4]  = {4, 4, 4, 1};

  const int TB = 256;
  auto blocks = [](long long n, int tb) { return (unsigned)((n + tb - 1) / tb); };

  for (int li = 0; li < 4; ++li) {
    const int fi = fis[li];
    const int H  = Hs[li];
    const int fo = H * 32;
    const int hShift = (H == 4) ? 2 : 0;
    const int fShift = (fo == 128) ? 7 : 5;
    const int nkb = (fi + 31) / 32;        // 3 for layer 0, else 4
    const int Kp  = nkb * 32;

    const float* lp[8];
    for (int j = 0; j < 8; ++j) lp[j] = (const float*)d_in[2 + li * 8 + j];
    const float *W = lp[0], *a_src = lp[1], *a_dst = lp[2], *bias = lp[3];
    const float *gamma = lp[4], *beta = lp[5], *rm = lp[6], *rv = lp[7];

    const float* xin = (li == 0) ? x0 : xbuf;

    // 0) pack operands into WMMA fragment order (bf16, zero-padded K)
    int aTotal = N * Kp;
    pack_a<<<blocks(aTotal, TB), TB, 0, stream>>>(xin, Ap, fi, nkb, aTotal);
    int bTotal = fo * Kp;
    pack_b<<<blocks(bTotal, TB), TB, 0, stream>>>(W, Bp, fi, fo, nkb, bTotal);

    // 1) hx = xin @ W  (WMMA bf16, one wave per 16x16 tile, fo/16 waves/block)
    if (nkb == 3)
      gemm_packed<3><<<N / 16, (fo / 16) * 32, 0, stream>>>(Ap, Bp, hx, fo);
    else
      gemm_packed<4><<<N / 16, (fo / 16) * 32, 0, stream>>>(Ap, Bp, hx, fo);

    // 2) attention logits per (node, head)
    attn_scores<<<blocks((long long)N << hShift, TB), TB, 0, stream>>>(
        hx, a_src, a_dst, as_, ad_, N, hShift);

    // 3) init accumulators: outb = 0, denom = 0, menc = 0 (== -max in enc order)
    fill_u32<<<blocks((long long)N * fo, TB), TB, 0, stream>>>((unsigned*)outb, 0u, N * fo);
    fill_u32<<<blocks((long long)N * H, TB), TB, 0, stream>>>((unsigned*)denom, 0u, N * H);
    fill_u32<<<blocks((long long)N * H, TB), TB, 0, stream>>>(menc, 0u, N * H);

    // 4) segment max
    edge_max<<<blocks((long long)(E + N) << hShift, TB), TB, 0, stream>>>(
        src, dst, as_, ad_, menc, E, N, hShift);

    // 5) exp + denom
    edge_exp<<<blocks((long long)(E + N) << hShift, TB), TB, 0, stream>>>(
        src, dst, as_, ad_, menc, exbuf, denom, E, N, hShift);

    // 6) normalize alphas once per (edge, head)
    edge_alpha<<<blocks((long long)(E + N) << hShift, TB), TB, 0, stream>>>(
        dst, exbuf, denom, E, N, hShift);

    // 7) weighted scatter-aggregate (float4 per thread)
    edge_agg<<<blocks((long long)(E + N) << (fShift - 2), TB), TB, 0, stream>>>(
        src, dst, exbuf, hx, outb, E, N, fShift);

    // 8) bias + BN (+ReLU); last layer writes d_out
    float* xdst = (li < 3) ? xbuf : (float*)d_out;
    finalize<<<blocks((long long)N << fShift, TB), TB, 0, stream>>>(
        outb, bias, gamma, beta, rm, rv, xdst, N, fShift, li < 3 ? 1 : 0);
  }
}
